// SmalldeckClassificationFlat_37434934952728
// MI455X (gfx1250) — compile-verified, matrix operand order checked by reading, not run on hardware
//
#include <hip/hip_runtime.h>
#include <hip/hip_bf16.h>

typedef __attribute__((ext_vector_type(16))) _Float16 v16h;
typedef __attribute__((ext_vector_type(8)))  _Float16 v8h;
typedef __attribute__((ext_vector_type(8)))  float    v8f;

// hero pair / board triple index tables (itertools.combinations order)
__constant__ int dHP[6][2]  = {{0,1},{0,2},{0,3},{1,2},{1,3},{2,3}};
__constant__ int dBT[10][3] = {{0,1,2},{0,1,3},{0,1,4},{0,2,3},{0,2,4},
                               {0,3,4},{1,2,3},{1,2,4},{1,3,4},{2,3,4}};

// ---------------------------------------------------------------------------
// Decode: sort hero(0..3)/board(4..8) by (rank,suit) asc (== stable argsort by
// suit then rank), emit card id (r-1)*s.
// ---------------------------------------------------------------------------
__global__ __launch_bounds__(256) void k_decode(const int* __restrict__ x,
                                                int* __restrict__ cardIdx, int M) {
  int n = blockIdx.x * blockDim.x + threadIdx.x;
  if (n >= M) return;
  int r[9], s[9];
#pragma unroll
  for (int i = 0; i < 9; ++i) { r[i] = x[n * 18 + 2 * i]; s[i] = x[n * 18 + 2 * i + 1]; }
  for (int i = 1; i < 4; ++i) {            // hero insertion sort
    int kr = r[i], ks = s[i], key = kr * 8 + ks, j = i - 1;
    while (j >= 0 && (r[j] * 8 + s[j]) > key) { r[j + 1] = r[j]; s[j + 1] = s[j]; --j; }
    r[j + 1] = kr; s[j + 1] = ks;
  }
  for (int i = 5; i < 9; ++i) {            // board insertion sort
    int kr = r[i], ks = s[i], key = kr * 8 + ks, j = i - 1;
    while (j >= 4 && (r[j] * 8 + s[j]) > key) { r[j + 1] = r[j]; s[j + 1] = s[j]; --j; }
    r[j + 1] = kr; s[j + 1] = ks;
  }
#pragma unroll
  for (int i = 0; i < 9; ++i) cardIdx[n * 9 + i] = (r[i] - 1) * s[i];
}

// ---------------------------------------------------------------------------
// Gather embeddings -> f16 MLP inputs. hero: [M*6,128], board: [M*10,192]
// ---------------------------------------------------------------------------
__global__ __launch_bounds__(256) void k_gather_hero(const float* __restrict__ emb,
                                                     const int* __restrict__ cardIdx,
                                                     _Float16* __restrict__ heroIn, int M) {
  long t = (long)blockIdx.x * blockDim.x + threadIdx.x;
  if (t >= (long)M * 768) return;
  int n = (int)(t / 768), rest = (int)(t % 768);
  int p = rest >> 7, k = rest & 127;
  int pos = (k < 64) ? dHP[p][0] : dHP[p][1];
  int card = cardIdx[n * 9 + pos];
  heroIn[t] = (_Float16)emb[card * 64 + (k & 63)];
}

__global__ __launch_bounds__(256) void k_gather_board(const float* __restrict__ emb,
                                                      const int* __restrict__ cardIdx,
                                                      _Float16* __restrict__ boardIn, int M) {
  long t = (long)blockIdx.x * blockDim.x + threadIdx.x;
  if (t >= (long)M * 1920) return;
  int n = (int)(t / 1920), rest = (int)(t % 1920);
  int b = rest / 192, k = rest % 192;
  int pos = 4 + dBT[b][k >> 6];
  int card = cardIdx[n * 9 + pos];
  boardIn[t] = (_Float16)emb[card * 64 + (k & 63)];
}

// ---------------------------------------------------------------------------
// Weight transpose + f16 convert: W[I][O] fp32 -> Wt[Opad][I] f16 (zero pad)
// ---------------------------------------------------------------------------
__global__ __launch_bounds__(256) void k_transpose_w(const float* __restrict__ W,
                                                     _Float16* __restrict__ Wt,
                                                     int I, int O, int Opad) {
  long t = (long)blockIdx.x * blockDim.x + threadIdx.x;
  if (t >= (long)Opad * I) return;
  int o = (int)(t / I), i = (int)(t % I);
  Wt[t] = (o < O) ? (_Float16)W[(long)i * O + o] : (_Float16)0.f;
}

__global__ __launch_bounds__(256) void k_pad_bias(const float* __restrict__ b,
                                                  float* __restrict__ bp, int O, int Opad) {
  int t = blockIdx.x * blockDim.x + threadIdx.x;
  if (t < Opad) bp[t] = (t < O) ? b[t] : 0.f;
}

// ---------------------------------------------------------------------------
// Fold oW1 (15360x512 fp32) into W1' transposed: W1t[512][2048] f16.
// hero cols c<768: sum over 10 board triples; board cols: sum over 6 pairs.
// ---------------------------------------------------------------------------
__global__ __launch_bounds__(256) void k_fold_oW1(const float* __restrict__ oW1,
                                                  _Float16* __restrict__ W1t) {
  int t = blockIdx.x * blockDim.x + threadIdx.x;  // 512*2048
  if (t >= 512 * 2048) return;
  int j = t % 512;      // consecutive threads -> consecutive oW1 cols (coalesced)
  int c = t / 512;
  float acc = 0.f;
  if (c < 768) {
    int p = c >> 7, k = c & 127;
    for (int b = 0; b < 10; ++b)
      acc += oW1[(size_t)((p * 10 + b) * 256 + k) * 512 + j];
  } else {
    int cc = c - 768, b = cc >> 7, k = cc & 127;
    for (int p = 0; p < 6; ++p)
      acc += oW1[(size_t)((p * 10 + b) * 256 + 128 + k) * 512 + j];
  }
  W1t[(size_t)j * 2048 + c] = (_Float16)acc;
}

// ---------------------------------------------------------------------------
// Pack per-hand features: A2[n][0:768]=heroFeat rows, [768:2048]=boardFeat rows
// ---------------------------------------------------------------------------
__global__ __launch_bounds__(256) void k_pack_A2(const _Float16* __restrict__ heroFeat,
                                                 const _Float16* __restrict__ boardFeat,
                                                 _Float16* __restrict__ A2, int M) {
  long t = (long)blockIdx.x * blockDim.x + threadIdx.x;
  if (t >= (long)M * 2048) return;
  int n = (int)(t >> 11), c = (int)(t & 2047);
  A2[t] = (c < 768) ? heroFeat[(long)n * 768 + c] : boardFeat[(long)n * 1280 + (c - 768)];
}

// ---------------------------------------------------------------------------
// WMMA GEMM: C[M][N] = lrelu(A[M][K] @ Bt^T + bias), f16 in/out, f32 acc.
// Bt is N x K row-major (pre-transposed weights). M mult 64, N mult 128, K mult 32.
// Block tile 64x128, 256 threads = 8 waves in 2(row) x 4(col); each wave owns a
// 32x32 sub-tile = four 16x16 f32 accumulators -> 4 v_wmma per K=32 step.
// Tiles are staged via CDNA5 async copy: global_load_async_to_lds_b128
// (ASYNCcnt-tracked, no VGPR round-trip); each wave s_wait_asynccnt 0 before the
// publish barrier. Next K tile is prefetched with global_prefetch_b8.
// ---------------------------------------------------------------------------
__global__ __launch_bounds__(256) void k_gemm_f16(const _Float16* __restrict__ A,
                                                  const _Float16* __restrict__ Bt,
                                                  const float* __restrict__ bias,
                                                  _Float16* __restrict__ C,
                                                  int M, int N, int K) {
  __shared__ __align__(16) _Float16 As[64][40];    // 40-half stride: 16B-aligned, bank-skewed
  __shared__ __align__(16) _Float16 Bs[128][40];

  const int tid  = threadIdx.x;
  const int lane = tid & 31;
  const int wave = tid >> 5;
  const int bm = blockIdx.x * 64;
  const int bn = blockIdx.y * 128;
  const int tr = (wave & 1) * 32;   // wave's 32-row strip in tile
  const int tc = (wave >> 1) * 32;  // wave's 32-col strip in tile

  const int lrow = tid >> 2;        // 0..63 : staging row
  const int lk   = (tid & 3) * 8;   // 0,8,16,24 : 8-half chunk

  const int hl  = lane >> 4;        // half-wave select
  const int l16 = lane & 15;
  const int kb  = hl * 8;

  // per-thread async staging addresses (LDS side is fixed; global side advances)
  const unsigned lds_a  = (unsigned)(uintptr_t)&As[lrow][lk];
  const unsigned lds_b0 = (unsigned)(uintptr_t)&Bs[lrow][lk];
  const unsigned lds_b1 = (unsigned)(uintptr_t)&Bs[64 + lrow][lk];
  const _Float16* ga  = &A [(size_t)(bm + lrow) * K + lk];
  const _Float16* gb0 = &Bt[(size_t)(bn + lrow) * K + lk];
  const _Float16* gb1 = &Bt[(size_t)(bn + 64 + lrow) * K + lk];

  v8f acc00 = {}, acc01 = {}, acc10 = {}, acc11 = {};

  for (int k0 = 0; k0 < K; k0 += 32) {
    // async copy global tile -> LDS (ASYNCcnt), 16B per lane per instruction
    asm volatile("global_load_async_to_lds_b128 %0, %1, off"
                 :: "v"(lds_a), "v"(ga) : "memory");
    asm volatile("global_load_async_to_lds_b128 %0, %1, off"
                 :: "v"(lds_b0), "v"(gb0) : "memory");
    asm volatile("global_load_async_to_lds_b128 %0, %1, off"
                 :: "v"(lds_b1), "v"(gb1) : "memory");
    if (k0 + 32 < K) {   // prefetch next K tile -> global_prefetch_b8
      __builtin_prefetch(ga + 32, 0, 0);
      __builtin_prefetch(gb0 + 32, 0, 0);
      __builtin_prefetch(gb1 + 32, 0, 0);
    }
    asm volatile("s_wait_asynccnt 0x0" ::: "memory");  // own copies landed
    __syncthreads();                                   // tile published

    v8h t0, t1;
    t0 = *(const v8h*)&As[tr + l16][kb];
    t1 = *(const v8h*)&As[tr + l16][kb + 16];
    v16h a0 = __builtin_shufflevector(t0, t1, 0,1,2,3,4,5,6,7,8,9,10,11,12,13,14,15);
    t0 = *(const v8h*)&As[tr + 16 + l16][kb];
    t1 = *(const v8h*)&As[tr + 16 + l16][kb + 16];
    v16h a1 = __builtin_shufflevector(t0, t1, 0,1,2,3,4,5,6,7,8,9,10,11,12,13,14,15);
    t0 = *(const v8h*)&Bs[tc + l16][kb];
    t1 = *(const v8h*)&Bs[tc + l16][kb + 16];
    v16h b0 = __builtin_shufflevector(t0, t1, 0,1,2,3,4,5,6,7,8,9,10,11,12,13,14,15);
    t0 = *(const v8h*)&Bs[tc + 16 + l16][kb];
    t1 = *(const v8h*)&Bs[tc + 16 + l16][kb + 16];
    v16h b1 = __builtin_shufflevector(t0, t1, 0,1,2,3,4,5,6,7,8,9,10,11,12,13,14,15);

    acc00 = __builtin_amdgcn_wmma_f32_16x16x32_f16(false, a0, false, b0, (short)0, acc00, false, false);
    acc01 = __builtin_amdgcn_wmma_f32_16x16x32_f16(false, a0, false, b1, (short)0, acc01, false, false);
    acc10 = __builtin_amdgcn_wmma_f32_16x16x32_f16(false, a1, false, b0, (short)0, acc10, false, false);
    acc11 = __builtin_amdgcn_wmma_f32_16x16x32_f16(false, a1, false, b1, (short)0, acc11, false, false);

    __syncthreads();                                   // safe to overwrite tile
    ga += 32; gb0 += 32; gb1 += 32;
  }

  // C/D layout: VGPR r, lane -> row = r + 8*(lane>>4), col = lane&15
  const int c0 = bn + tc + l16;
  const int c1 = c0 + 16;
  const float bia0 = bias[c0];
  const float bia1 = bias[c1];
#pragma unroll
  for (int r = 0; r < 8; ++r) {
    int row0 = bm + tr + r + hl * 8;
    int row1 = row0 + 16;
    float v;
    v = acc00[r] + bia0; v = fmaxf(v, 0.01f * v); C[(size_t)row0 * N + c0] = (_Float16)v;
    v = acc01[r] + bia1; v = fmaxf(v, 0.01f * v); C[(size_t)row0 * N + c1] = (_Float16)v;
    v = acc10[r] + bia0; v = fmaxf(v, 0.01f * v); C[(size_t)row1 * N + c0] = (_Float16)v;
    v = acc11[r] + bia1; v = fmaxf(v, 0.01f * v); C[(size_t)row1 * N + c1] = (_Float16)v;
  }
}

// ---------------------------------------------------------------------------
// Final head: out[n][a] = sb[a] + sum_{k<127} r3[n][k] * sW[k][a]  (fp32 out)
// ---------------------------------------------------------------------------
__global__ __launch_bounds__(256) void k_final(const _Float16* __restrict__ r3,
                                               const float* __restrict__ sW,
                                               const float* __restrict__ sb,
                                               float* __restrict__ out, int M) {
  int t = blockIdx.x * blockDim.x + threadIdx.x;
  if (t >= M * 9) return;
  int n = t / 9, a = t % 9;
  float acc = sb[a];
  for (int k = 0; k < 127; ++k)
    acc += (float)r3[(size_t)n * 128 + k] * sW[k * 9 + a];
  out[t] = acc;
}

// ---------------------------------------------------------------------------
extern "C" void kernel_launch(void* const* d_in, const int* in_sizes, int n_in,
                              void* d_out, int out_size, void* d_ws, size_t ws_size,
                              hipStream_t stream) {
  const int*   x    = (const int*)  d_in[0];
  const float* emb  = (const float*)d_in[1];
  const float* hW1  = (const float*)d_in[2];
  const float* hb1  = (const float*)d_in[3];
  const float* hW2  = (const float*)d_in[4];
  const float* hb2  = (const float*)d_in[5];
  const float* bW1  = (const float*)d_in[6];
  const float* bb1  = (const float*)d_in[7];
  const float* bW2  = (const float*)d_in[8];
  const float* bb2  = (const float*)d_in[9];
  const float* oW1  = (const float*)d_in[10];
  const float* ob1  = (const float*)d_in[11];
  const float* oW2  = (const float*)d_in[12];
  const float* ob2  = (const float*)d_in[13];
  const float* oW3  = (const float*)d_in[14];
  const float* ob3  = (const float*)d_in[15];
  const float* sW   = (const float*)d_in[16];
  const float* sb   = (const float*)d_in[17];
  float* out = (float*)d_out;

  const int M  = in_sizes[0] / 18;   // 2048
  const int MH = M * 6;
  const int MB = M * 10;

  // workspace carve-up (256B aligned)
  uint8_t* ws = (uint8_t*)d_ws;
  size_t off = 0;
  auto alloc = [&](size_t bytes) -> void* {
    off = (off + 255) & ~(size_t)255;
    void* p = ws + off;
    off += bytes;
    return p;
  };
  int*       cardIdx   = (int*)      alloc((size_t)M * 9 * 4);
  _Float16*  heroIn    = (_Float16*) alloc((size_t)MH * 128 * 2);
  _Float16*  boardIn   = (_Float16*) alloc((size_t)MB * 192 * 2);
  _Float16*  hW1t      = (_Float16*) alloc((size_t)512 * 128 * 2);
  _Float16*  hW2t      = (_Float16*) alloc((size_t)128 * 512 * 2);
  _Float16*  bW1t      = (_Float16*) alloc((size_t)512 * 192 * 2);
  _Float16*  bW2t      = (_Float16*) alloc((size_t)128 * 512 * 2);
  _Float16*  W1t       = (_Float16*) alloc((size_t)512 * 2048 * 2);
  _Float16*  oW2t      = (_Float16*) alloc((size_t)256 * 512 * 2);
  _Float16*  oW3t      = (_Float16*) alloc((size_t)128 * 256 * 2);
  float*     ob3p      = (float*)    alloc(128 * 4);
  _Float16*  heroHid   = (_Float16*) alloc((size_t)MH * 512 * 2);
  _Float16*  heroFeat  = (_Float16*) alloc((size_t)MH * 128 * 2);
  _Float16*  boardHid  = (_Float16*) alloc((size_t)MB * 512 * 2);
  _Float16*  boardFeat = (_Float16*) alloc((size_t)MB * 128 * 2);
  _Float16*  A2        = (_Float16*) alloc((size_t)M * 2048 * 2);
  _Float16*  r1        = (_Float16*) alloc((size_t)M * 512 * 2);
  _Float16*  r2        = (_Float16*) alloc((size_t)M * 256 * 2);
  _Float16*  r3        = (_Float16*) alloc((size_t)M * 128 * 2);
  (void)ws_size; (void)n_in; (void)out_size;

  auto cdiv = [](long a, long b) { return (int)((a + b - 1) / b); };

  // --- weight prep (f16 transposed) ---
  k_transpose_w<<<cdiv(512L * 128, 256), 256, 0, stream>>>(hW1, hW1t, 128, 512, 512);
  k_transpose_w<<<cdiv(128L * 512, 256), 256, 0, stream>>>(hW2, hW2t, 512, 128, 128);
  k_transpose_w<<<cdiv(512L * 192, 256), 256, 0, stream>>>(bW1, bW1t, 192, 512, 512);
  k_transpose_w<<<cdiv(128L * 512, 256), 256, 0, stream>>>(bW2, bW2t, 512, 128, 128);
  k_transpose_w<<<cdiv(256L * 512, 256), 256, 0, stream>>>(oW2, oW2t, 512, 256, 256);
  k_transpose_w<<<cdiv(128L * 256, 256), 256, 0, stream>>>(oW3, oW3t, 256, 127, 128);
  k_pad_bias<<<1, 256, 0, stream>>>(ob3, ob3p, 127, 128);
  k_fold_oW1<<<cdiv(512L * 2048, 256), 256, 0, stream>>>(oW1, W1t);

  // --- decode + gather ---
  k_decode<<<cdiv(M, 256), 256, 0, stream>>>(x, cardIdx, M);
  k_gather_hero<<<cdiv((long)M * 768, 256), 256, 0, stream>>>(emb, cardIdx, heroIn, M);
  k_gather_board<<<cdiv((long)M * 1920, 256), 256, 0, stream>>>(emb, cardIdx, boardIn, M);

  // --- hero / board MLPs (dedup: 6 and 10 unique rows per hand) ---
  k_gemm_f16<<<dim3(MH / 64, 4), 256, 0, stream>>>(heroIn,  hW1t, hb1, heroHid,  MH, 512, 128);
  k_gemm_f16<<<dim3(MH / 64, 1), 256, 0, stream>>>(heroHid, hW2t, hb2, heroFeat, MH, 128, 512);
  k_gemm_f16<<<dim3(MB / 64, 4), 256, 0, stream>>>(boardIn, bW1t, bb1, boardHid, MB, 512, 192);
  k_gemm_f16<<<dim3(MB / 64, 1), 256, 0, stream>>>(boardHid, bW2t, bb2, boardFeat, MB, 128, 512);

  // --- folded output layer 1 + MLP tail ---
  k_pack_A2<<<cdiv((long)M * 2048, 256), 256, 0, stream>>>(heroFeat, boardFeat, A2, M);
  k_gemm_f16<<<dim3(M / 64, 4), 256, 0, stream>>>(A2, W1t,  ob1,  r1, M, 512, 2048);
  k_gemm_f16<<<dim3(M / 64, 2), 256, 0, stream>>>(r1, oW2t, ob2,  r2, M, 256, 512);
  k_gemm_f16<<<dim3(M / 64, 1), 256, 0, stream>>>(r2, oW3t, ob3p, r3, M, 128, 256);

  // --- head ---
  k_final<<<cdiv((long)M * 9, 256), 256, 0, stream>>>(r3, sW, sb, out, M);
}
